// Quantize_14826227106282
// MI455X (gfx1250) — compile-verified, matrix operand order checked by reading, not run on hardware
//
#include <hip/hip_runtime.h>
#include <hip/hip_bf16.h>
#include <math.h>

#define B_ 32768
#define D_ 256
#define K_ 2048

typedef __attribute__((ext_vector_type(16))) __bf16       v16bf;
typedef __attribute__((ext_vector_type(8)))  float        v8f;
typedef __attribute__((ext_vector_type(8)))  unsigned int v8u;

// Offset of bf16 element e (0..15) within a 16-bit 16x32 A/B WMMA fragment,
// g = lane>>4 (which 16-lane half). Matches CDNA5 ISA layout:
// lanes 0-15: V0-3 -> K=0..7, V4-7 -> K=16..23; lanes 16-31: +8.
__device__ __forceinline__ int frag_off(int e, int g) {
  int v = e >> 1;
  return ((v & 4) ? 16 : 0) + g * 8 + (v & 3) * 2 + (e & 1);
}

__device__ __forceinline__ float wave_sum32(float v) {
#pragma unroll
  for (int s = 16; s > 0; s >>= 1) v += __shfl_xor(v, s, 32);
  return v;
}

// --- CDNA5 async global->LDS copy (ASYNCcnt-tracked DMA path) --------------
__device__ __forceinline__ void async_ld_b128(unsigned lds_byte_addr,
                                              const char* gptr) {
  // VDST = LDS byte address VGPR, VADDR = 64-bit global address (GV mode).
  asm volatile("global_load_async_to_lds_b128 %0, %1, off"
               :: "v"(lds_byte_addr), "v"(gptr)
               : "memory");
}
__device__ __forceinline__ void wait_async_le4() {
  asm volatile("s_wait_asynccnt 0x4" ::: "memory");  // allow next batch in flight
}
__device__ __forceinline__ void wait_async_0() {
  asm volatile("s_wait_asynccnt 0x0" ::: "memory");
}

// ---------------------------------------------------------------------------
// Kernel 1: codebook -> bf16 hi/lo split + per-code squared norm.
// One wave per code row (K=2048 rows, 8 waves/block).
// ---------------------------------------------------------------------------
__global__ void __launch_bounds__(256) vq_prep(const float* __restrict__ cb,
                                               __bf16* __restrict__ ch,
                                               __bf16* __restrict__ cl,
                                               float* __restrict__ cnorm) {
  const int wid = threadIdx.x >> 5, lane = threadIdx.x & 31;
  const int row = blockIdx.x * 8 + wid;
  if (row >= K_) return;
  const float* cr = cb + (size_t)row * D_;
  float ss = 0.f;
#pragma unroll
  for (int j = 0; j < D_ / 32; ++j) {
    const int d = lane + 32 * j;
    const float v = cr[d];
    const __bf16 h = (__bf16)v;
    ch[(size_t)row * D_ + d] = h;
    cl[(size_t)row * D_ + d] = (__bf16)(v - (float)h);
    ss += v * v;
  }
  ss = wave_sum32(ss);
  if (lane == 0) cnorm[row] = ss;
}

// ---------------------------------------------------------------------------
// Kernel 2: fused distance GEMM (bf16x3 WMMA) + running argmin.
// Each wave owns a 16-row x tile (VGPR-resident, hi/lo bf16 split). The block
// sweeps all K=2048 codes in 16-column chunks; each chunk (8KB ch + 8KB cl)
// is DMA'd into LDS once per block with global_load_async_to_lds_b128
// (double-buffered, shared by 8 waves -> 8x less global traffic), then each
// wave runs 8 k-steps x 3 independent v_wmma_f32_16x16x32_bf16.
// argmin score = ||c||^2 - 2*dot (x-norm is constant per row).
// ---------------------------------------------------------------------------
__global__ void __launch_bounds__(256)
vq_argmin(const float* __restrict__ x,
          const __bf16* __restrict__ ch, const __bf16* __restrict__ cl,
          const float* __restrict__ cnorm, int* __restrict__ ids) {
  const int tid = threadIdx.x;
  const int wid = tid >> 5, lane = tid & 31;
  const int g = lane >> 4, ln = lane & 15;
  const int m0 = (blockIdx.x * 8 + wid) * 16;

  // [buf][h=0/l=1][16 codes x 256 bf16 = 8KB]
  __shared__ __align__(16) unsigned char smem[2][2][8192];

  const char* chb = (const char*)ch;
  const char* clb = (const char*)cl;
  const int o0 = tid * 16, o1 = 4096 + tid * 16;  // 64B per thread per chunk

  // Prologue: DMA chunk 0 into buffer 0 (4 async b128 per thread).
  {
    unsigned bh = (unsigned)(unsigned long long)(void*)&smem[0][0][0];
    unsigned bl = (unsigned)(unsigned long long)(void*)&smem[0][1][0];
    async_ld_b128(bh + o0, chb + o0);
    async_ld_b128(bh + o1, chb + o1);
    async_ld_b128(bl + o0, clb + o0);
    async_ld_b128(bl + o1, clb + o1);
  }

  // Preload x tile as A fragments, split into bf16 hi + residual lo
  // (overlaps with the chunk-0 DMA).
  v16bf Ah[8], Al[8];
  const float* xr = x + (size_t)(m0 + ln) * D_;
#pragma unroll
  for (int kk = 0; kk < 8; ++kk) {
#pragma unroll
    for (int e = 0; e < 16; ++e) {
      const int d = kk * 32 + frag_off(e, g);
      const float v = xr[d];
      const __bf16 h = (__bf16)v;
      Ah[kk][e] = h;
      Al[kk][e] = (__bf16)(v - (float)h);
    }
  }

  float minv[8];
  int   minid[8];
#pragma unroll
  for (int r = 0; r < 8; ++r) { minv[r] = 3.4e38f; minid[r] = 0; }

  for (int nc = 0; nc < K_ / 16; ++nc) {
    const int buf = nc & 1;
    // Prefetch next chunk into the other buffer before waiting (its target
    // buffer was released by the trailing barrier of iteration nc-1).
    if (nc + 1 < K_ / 16) {
      const size_t nb = (size_t)(nc + 1) * 8192;
      unsigned bh = (unsigned)(unsigned long long)(void*)&smem[buf ^ 1][0][0];
      unsigned bl = (unsigned)(unsigned long long)(void*)&smem[buf ^ 1][1][0];
      async_ld_b128(bh + o0, chb + nb + o0);
      async_ld_b128(bh + o1, chb + nb + o1);
      async_ld_b128(bl + o0, clb + nb + o0);
      async_ld_b128(bl + o1, clb + nb + o1);
      wait_async_le4();   // batch nc done; batch nc+1 still in flight
    } else {
      wait_async_0();     // final chunk: drain
    }
    __syncthreads();      // all waves' DMA for chunk nc visible

    const int n0 = nc * 16;
    const int col = n0 + ln;               // this lane's column (code index)
    v8f acc0 = {}, acc1 = {}, acc2 = {};   // independent: no WMMA RAW hazard
    const uint4* lch = (const uint4*)&smem[buf][0][0];
    const uint4* lcl = (const uint4*)&smem[buf][1][0];
#pragma unroll
    for (int kk = 0; kk < 8; ++kk) {
      // dword base = ln*128 + kk*16; fragment = dwords {g*4..g*4+3, 8+g*4..}
      const int u4 = ln * 32 + kk * 4;
      const uint4 h0 = lch[u4 + g];
      const uint4 h1 = lch[u4 + 2 + g];
      const uint4 l0 = lcl[u4 + g];
      const uint4 l1 = lcl[u4 + 2 + g];
      const v8u bh = {h0.x, h0.y, h0.z, h0.w, h1.x, h1.y, h1.z, h1.w};
      const v8u bl = {l0.x, l0.y, l0.z, l0.w, l1.x, l1.y, l1.z, l1.w};
      const v16bf Bh = __builtin_bit_cast(v16bf, bh);
      const v16bf Bl = __builtin_bit_cast(v16bf, bl);
      acc0 = __builtin_amdgcn_wmma_f32_16x16x32_bf16(false, Ah[kk], false, Bh,
                                                     (short)0, acc0, false, false);
      acc1 = __builtin_amdgcn_wmma_f32_16x16x32_bf16(false, Ah[kk], false, Bl,
                                                     (short)0, acc1, false, false);
      acc2 = __builtin_amdgcn_wmma_f32_16x16x32_bf16(false, Al[kk], false, Bh,
                                                     (short)0, acc2, false, false);
    }
    const float cn = cnorm[col];
#pragma unroll
    for (int r = 0; r < 8; ++r) {
      // D layout: VGPR r, this lane -> row m0 + g*8 + r, column `col`.
      const float s = cn - 2.f * (acc0[r] + acc1[r] + acc2[r]);
      if (s < minv[r]) { minv[r] = s; minid[r] = col; }
    }
    __syncthreads();      // all waves done reading buf before it is re-filled
  }

  // Min-reduce across the 16 lanes of each half-wave (xor masks < 16 stay
  // inside the half). Tie-break on lower index to match jnp.argmin.
#pragma unroll
  for (int sft = 1; sft < 16; sft <<= 1) {
#pragma unroll
    for (int r = 0; r < 8; ++r) {
      const float ov = __shfl_xor(minv[r], sft, 32);
      const int   oi = __shfl_xor(minid[r], sft, 32);
      if (ov < minv[r] || (ov == minv[r] && oi < minid[r])) {
        minv[r] = ov; minid[r] = oi;
      }
    }
  }
  if (ln == 0) {
#pragma unroll
    for (int r = 0; r < 8; ++r) ids[m0 + g * 8 + r] = minid[r];
  }
}

// ---------------------------------------------------------------------------
// Kernel 3: gather + rotation trick + loss. One wave per row, 8 elems/lane.
// Outputs concatenated: emb_out [B*D], ids [B] (as float), loss [B].
// ---------------------------------------------------------------------------
__global__ void __launch_bounds__(256)
vq_rotate(const float* __restrict__ x, const float* __restrict__ cb,
          const int* __restrict__ ids, float* __restrict__ out) {
  const int wid = threadIdx.x >> 5, lane = threadIdx.x & 31;
  const int row = blockIdx.x * 8 + wid;
  if (row >= B_) return;
  const float* xr = x + (size_t)row * D_;
  const int id = ids[row];
  const float* er = cb + (size_t)id * D_;

  float xs[8], es[8];
  float sx = 0.f, se = 0.f;
#pragma unroll
  for (int j = 0; j < 8; ++j) {
    const int d = lane + 32 * j;
    xs[j] = xr[d]; es[j] = er[d];
    sx += xs[j] * xs[j]; se += es[j] * es[j];
  }
  sx = wave_sum32(sx); se = wave_sum32(se);
  const float xn = sqrtf(sx), en = sqrtf(se);
  const float iu = 1.f / (xn + 1e-8f), iq = 1.f / (en + 1e-8f);

  float us[8], qs[8];
  float swr = 0.f, d2 = 0.f;
#pragma unroll
  for (int j = 0; j < 8; ++j) {
    us[j] = xs[j] * iu; qs[j] = es[j] * iq;
    const float wr = us[j] + qs[j];
    swr += wr * wr;
    const float df = xs[j] - es[j];
    d2 += df * df;
  }
  swr = wave_sum32(swr); d2 = wave_sum32(d2);
  const float iw = 1.f / fmaxf(sqrtf(swr), 1e-6f);

  float ew = 0.f, eu = 0.f;
#pragma unroll
  for (int j = 0; j < 8; ++j) {
    const float wv = (us[j] + qs[j]) * iw;
    ew += xs[j] * wv; eu += xs[j] * us[j];
  }
  ew = wave_sum32(ew); eu = wave_sum32(eu);
  const float scale = en / (xn + 1e-6f);

#pragma unroll
  for (int j = 0; j < 8; ++j) {
    const int d = lane + 32 * j;
    const float wv = (us[j] + qs[j]) * iw;
    out[(size_t)row * D_ + d] = (xs[j] - 2.f * ew * wv + 2.f * eu * qs[j]) * scale;
  }
  if (lane == 0) {
    out[(size_t)B_ * D_ + row] = (float)id;          // ids
    out[(size_t)B_ * D_ + B_ + row] = 1.25f * d2;    // emb_loss + 0.25*query_loss
  }
}

// ---------------------------------------------------------------------------
extern "C" void kernel_launch(void* const* d_in, const int* in_sizes, int n_in,
                              void* d_out, int out_size, void* d_ws, size_t ws_size,
                              hipStream_t stream) {
  (void)in_sizes; (void)n_in; (void)out_size; (void)ws_size;
  const float* x  = (const float*)d_in[0];
  const float* cb = (const float*)d_in[1];
  float* out = (float*)d_out;

  // Workspace layout (~2.14 MB):
  char* w = (char*)d_ws;
  __bf16* ch   = (__bf16*)w;                         // K*D bf16 (1 MB)
  __bf16* cl   = ch + (size_t)K_ * D_;               // K*D bf16 (1 MB)
  float* cnorm = (float*)(cl + (size_t)K_ * D_);     // K floats (8 KB)
  int*   ids   = (int*)(cnorm + K_);                 // B ints (128 KB)

  vq_prep  <<<K_ / 8,   256, 0, stream>>>(cb, ch, cl, cnorm);
  vq_argmin<<<B_ / 128, 256, 0, stream>>>(x, ch, cl, cnorm, ids);
  vq_rotate<<<B_ / 8,   256, 0, stream>>>(x, cb, ids, out);
}
